// MultiHeadedAttention_62526133895682
// MI455X (gfx1250) — matrix operand order repaired and checked by reading, not compiled, not run on hardware
//
#include <hip/hip_runtime.h>

typedef __attribute__((ext_vector_type(16))) __bf16 v16bf;
typedef __attribute__((ext_vector_type(8)))  float  v8f;
typedef unsigned int u32x4 __attribute__((ext_vector_type(4)));
typedef int          i32x4 __attribute__((ext_vector_type(4)));
typedef int          i32x8 __attribute__((ext_vector_type(8)));

#define S_LEN   2048
#define D_MODEL 1024
#define NH      16
#define HD      64
#define BATCH   2
#define M_TOK   (BATCH * S_LEN)   // 4096

// workspace offsets in ushort (bf16-bit) units; total 24Mi ushorts = 48 MB
#define OFF_XH   (0u)
#define OFF_WQ   (4194304u)
#define OFF_WK   (5242880u)
#define OFF_WV   (6291456u)
#define OFF_WO   (7340032u)
#define OFF_Q    (8388608u)    // [b][h][s][hd], Q pre-scaled by 1/32
#define OFF_K    (12582912u)   // [b][h][s][hd]
#define OFF_VT   (16777216u)   // [b][h][hd][s]  (V transposed)
#define OFF_CTX  (20971520u)   // [b*s][1024]

union BfCvt { __bf16 b; unsigned short u; };
__device__ __forceinline__ unsigned short bfbits(float f) {
  BfCvt c; c.b = (__bf16)f; return c.u;       // native v_cvt f32->bf16
}

union FragU { uint4 q[2]; v16bf v; unsigned short s[16]; };
union Pack8 { uint4 u; unsigned short s[8]; };

// A-matrix 16x32 bf16 frag: lane holds row (lane&15); K = k + 16*hi + {0..15}
__device__ __forceinline__ v16bf load_a_frag(const unsigned short* row, int k, int hi) {
  FragU f;
  f.q[0] = *(const uint4*)(row + k + 16 * hi);
  f.q[1] = *(const uint4*)(row + k + 16 * hi + 8);
  return f.v;
}
// B-matrix 32x16 bf16 frag: lane holds column (lane&15); K = 16*hi + {0..15}
__device__ __forceinline__ v16bf load_b_frag(const unsigned short* col, int k, int hi) {
  FragU f;
  f.q[0] = *(const uint4*)(col + k + 16 * hi);
  f.q[1] = *(const uint4*)(col + k + 16 * hi + 8);
  return f.v;
}

__device__ __forceinline__ v8f wmma_bf16(v16bf a, v16bf b, v8f c) {
  return __builtin_amdgcn_wmma_f32_16x16x32_bf16(false, a, false, b, (short)0, c, false, false);
}

// ---- TDM: load a 64-row x 32-elem bf16 tile (row stride 1024 elems) to LDS ----
// D# per CDNA5 ISA: group0 {count=1, lds_addr, global_addr, type=2},
// group1 {data_size=2B, tensor 1024x1024, tile 32x64, stride0=1024}.
__device__ __forceinline__ void tdm_load_w_tile(const unsigned short* gsrc, unsigned ldsOff) {
  unsigned long long ga = (unsigned long long)gsrc;
  u32x4 g0;
  g0[0] = 1u;                                              // count=1 (valid descriptor)
  g0[1] = ldsOff;                                          // lds byte address
  g0[2] = (unsigned)ga;                                    // global_addr[31:0]
  g0[3] = ((unsigned)(ga >> 32) & 0x01FFFFFFu) | 0x80000000u; // addr[56:32] | type=2
  i32x8 g1;
  g1[0] = 0x00010000;        // workgroup_mask=0, data_size=1 (2 bytes)
  g1[1] = 0x04000000;        // tensor_dim0=1024 (low16 << 16)
  g1[2] = 0x04000000;        // tensor_dim0 hi | tensor_dim1=1024 (low16 << 16)
  g1[3] = 0x00200000;        // tensor_dim1 hi | tile_dim0=32 (<<16)
  g1[4] = 64;                // tile_dim1=64, tile_dim2=0
  g1[5] = 1024;              // tensor_dim0_stride = 1024
  g1[6] = 0;
  g1[7] = 0;
  i32x4 gz = {0, 0, 0, 0};
#if __has_include(<hip/amd_detail/amd_gfx1250_TDM.h>)
  i32x8 gz8 = {0, 0, 0, 0, 0, 0, 0, 0};
  __builtin_amdgcn_tensor_load_to_lds(g0, g1, gz, gz, gz8, 0);   // clang-23 6-arg form
#else
  __builtin_amdgcn_tensor_load_to_lds(g0, g1, gz, gz, 0);        // ROCm 7.2 5-arg form
#endif
}

// ---------------- kernel 1: fp32 -> bf16 ----------------
__global__ void convert_bf16(const float* __restrict__ src,
                             unsigned short* __restrict__ dst, int n) {
  int i = blockIdx.x * blockDim.x + threadIdx.x;
  int stride = gridDim.x * blockDim.x;
  for (; i < n; i += stride) dst[i] = bfbits(src[i]);
}

// ---------------- kernel 2: QKV GEMM (y = x @ W^T) ----------------
// grid (16, 16, 3), block 256 (8 waves). wave -> 32x64 output tile.
// Weight tile (64 cols x 32 k) staged to LDS by TDM, double-buffered.
__global__ __launch_bounds__(256) void qkv_gemm(
    const unsigned short* __restrict__ xh,
    const unsigned short* __restrict__ wq,
    const unsigned short* __restrict__ wk,
    const unsigned short* __restrict__ wv,
    unsigned short* __restrict__ qout,
    unsigned short* __restrict__ kout,
    unsigned short* __restrict__ vtout) {
  __shared__ __align__(16) unsigned short bsh[2][64 * 32];

  const int lane = threadIdx.x & 31;
  const int wave = threadIdx.x >> 5;
  const int lo = lane & 15, hi = lane >> 4;
  const int m0 = (blockIdx.x * 8 + wave) * 32;
  const int nb = blockIdx.y * 64;
  const int mode = blockIdx.z;
  const unsigned short* w = (mode == 0) ? wq : ((mode == 1) ? wk : wv);

  const unsigned short* arow0 = xh + (size_t)(m0 + lo) * D_MODEL;
  const unsigned short* arow1 = xh + (size_t)(m0 + 16 + lo) * D_MODEL;

  if (wave == 0)
    tdm_load_w_tile(w + (size_t)nb * D_MODEL, (unsigned)(uintptr_t)&bsh[0][0]);

  v8f acc[2][4] = {};
  for (int k = 0; k < D_MODEL; k += 32) {
    const int cur = (k >> 5) & 1;
    if (wave == 0) __builtin_amdgcn_s_wait_tensorcnt(0);
    __syncthreads();
    if (wave == 0 && (k + 32) < D_MODEL)
      tdm_load_w_tile(w + (size_t)nb * D_MODEL + (k + 32),
                      (unsigned)(uintptr_t)&bsh[cur ^ 1][0]);

    __builtin_prefetch(arow0 + k + 64, 0, 1);
    __builtin_prefetch(arow1 + k + 64, 0, 1);
    v16bf a0 = load_a_frag(arow0, k, hi);
    v16bf a1 = load_a_frag(arow1, k, hi);
#pragma unroll
    for (int j = 0; j < 4; ++j) {
      const unsigned short* bcol = &bsh[cur][(j * 16 + lo) * 32 + 16 * hi];
      FragU bf;
      bf.q[0] = *(const uint4*)(bcol);
      bf.q[1] = *(const uint4*)(bcol + 8);
      acc[0][j] = wmma_bf16(a0, bf.v, acc[0][j]);
      acc[1][j] = wmma_bf16(a1, bf.v, acc[1][j]);
    }
  }

  if (mode < 2) {
    const float scale = (mode == 0) ? 0.03125f : 1.0f;  // 1/sqrt(1024) folded into Q
    unsigned short* dst = (mode == 0) ? qout : kout;
#pragma unroll
    for (int mi = 0; mi < 2; ++mi) {
#pragma unroll
      for (int j = 0; j < 4; ++j) {
        int n = nb + j * 16 + lo;
        int h = n >> 6, hd = n & 63;
#pragma unroll
        for (int r = 0; r < 8; ++r) {
          int m = m0 + mi * 16 + r + 8 * hi;
          int b = m >> 11, s = m & 2047;
          dst[(((size_t)(b * NH + h)) * S_LEN + s) * HD + hd] = bfbits(acc[mi][j][r] * scale);
        }
      }
    }
  } else {
    // V transposed: Vt[b][h][hd][s]; C-frag rows contiguous per lane -> 16B store
#pragma unroll
    for (int mi = 0; mi < 2; ++mi) {
      int m = m0 + mi * 16 + 8 * hi;
      int b = m >> 11, s = m & 2047;
#pragma unroll
      for (int j = 0; j < 4; ++j) {
        int n = nb + j * 16 + lo;
        int h = n >> 6, hd = n & 63;
        Pack8 p;
#pragma unroll
        for (int r = 0; r < 8; ++r) p.s[r] = bfbits(acc[mi][j][r]);
        *(uint4*)(vtout + ((((size_t)(b * NH + h)) * HD + hd) * S_LEN + s)) = p.u;
      }
    }
  }
}

// ---------------- kernel 3: causal flash attention ----------------
// grid (32, 16, 2), block 128 (4 waves); wave handles 16 queries of one (b,h).
__global__ __launch_bounds__(128) void attn(
    const unsigned short* __restrict__ qh,
    const unsigned short* __restrict__ kh,
    const unsigned short* __restrict__ vt,
    unsigned short* __restrict__ ctx) {
  const int lane = threadIdx.x & 31;
  const int wave = threadIdx.x >> 5;
  const int lo = lane & 15, hi = lane >> 4;
  const int h = blockIdx.y, b = blockIdx.z;
  const int qb = blockIdx.x * 64 + wave * 16;

  const unsigned short* qbh = qh + ((size_t)(b * NH + h)) * S_LEN * HD;
  const unsigned short* kbh = kh + ((size_t)(b * NH + h)) * S_LEN * HD;
  const unsigned short* vbh = vt + ((size_t)(b * NH + h)) * HD * S_LEN;

  // Q as B-matrix of S^T = K * Q^T : lane column = query
  const unsigned short* qrow = qbh + (size_t)(qb + lo) * HD;
  const v16bf bq0 = load_b_frag(qrow, 0, hi);
  const v16bf bq1 = load_b_frag(qrow, 32, hi);

  v8f acc[4] = {};                 // ctx^T accumulators, 4 d-tiles of 16
  float Mrun = -INFINITY, Lrun = 0.0f;
  const int q = qb + lo;

  for (int kb = 0; kb < qb + 16; kb += 32) {
    const unsigned short* k0row = kbh + (size_t)(kb + lo) * HD;
    const unsigned short* k1row = kbh + (size_t)(kb + 16 + lo) * HD;
    v16bf ak0l = load_a_frag(k0row, 0, hi);
    v16bf ak0h = load_a_frag(k0row, 32, hi);
    v16bf ak1l = load_a_frag(k1row, 0, hi);
    v16bf ak1h = load_a_frag(k1row, 32, hi);

    v8f s0 = {}, s1 = {};
    s0 = wmma_bf16(ak0l, bq0, s0);
    s0 = wmma_bf16(ak0h, bq1, s0);
    s1 = wmma_bf16(ak1l, bq0, s1);
    s1 = wmma_bf16(ak1h, bq1, s1);

    // causal mask + online softmax (row = query = lane&15; pair (l, l^16) shares row)
    float p0[8], p1[8];
    float mloc = -INFINITY;
#pragma unroll
    for (int r = 0; r < 8; ++r) {
      int key0 = kb + r + 8 * hi;          // S^T C-frag: element r -> key row r+8*hi
      int key1 = key0 + 16;
      float v0 = (key0 <= q) ? s0[r] : -INFINITY;
      float v1 = (key1 <= q) ? s1[r] : -INFINITY;
      p0[r] = v0; p1[r] = v1;
      mloc = fmaxf(mloc, fmaxf(v0, v1));
    }
    mloc = fmaxf(mloc, __shfl_xor(mloc, 16, 32));
    float Mnew = fmaxf(Mrun, mloc);
    float alpha = __expf(Mrun - Mnew);
    float lsum = 0.0f;
#pragma unroll
    for (int r = 0; r < 8; ++r) {
      p0[r] = __expf(p0[r] - Mnew); lsum += p0[r];
      p1[r] = __expf(p1[r] - Mnew); lsum += p1[r];
    }
    lsum += __shfl_xor(lsum, 16, 32);
    Lrun = Lrun * alpha + lsum;
    Mrun = Mnew;

    // Build P^T B-frag (K = key, N = query). Lane needs keys kb+16*hi+{0..15}:
    //  low lanes: own p0 (keys 0..7)  + partner p0 (keys 8..15)
    //  high lanes: partner p1 (16..23) + own p1 (24..31)
    FragU pf;
#pragma unroll
    for (int r = 0; r < 8; ++r) {
      float o0 = __shfl_xor(p0[r], 16, 32);
      float o1 = __shfl_xor(p1[r], 16, 32);
      pf.s[r]     = bfbits(hi ? o1 : p0[r]);
      pf.s[8 + r] = bfbits(hi ? p1[r] : o0);
    }

    // ctx^T += V^T * P^T ; A = Vt d-tile (16 x 32 keys)
#pragma unroll
    for (int t = 0; t < 4; ++t) {
      const unsigned short* vrow = vbh + (size_t)(t * 16 + lo) * S_LEN + kb;
#pragma unroll
      for (int r = 0; r < 8; ++r) acc[t][r] *= alpha;
      v16bf av = load_a_frag(vrow, 0, hi);
      acc[t] = wmma_bf16(av, pf.v, acc[t]);
    }
  }

  const float inv = 1.0f / Lrun;
  const size_t tokrow = ((size_t)(b * S_LEN) + qb + lo) * D_MODEL;
#pragma unroll
  for (int t = 0; t < 4; ++t) {
    Pack8 p;
#pragma unroll
    for (int r = 0; r < 8; ++r) p.s[r] = bfbits(acc[t][r] * inv);
    *(uint4*)(ctx + tokrow + h * HD + t * 16 + 8 * hi) = p.u;
  }
}

// ---------------- kernel 4: out = ctx @ Wo^T + bo (fp32 out) ----------------
// grid (16, 16), block 256; same TDM-staged structure as qkv_gemm.
__global__ __launch_bounds__(256) void out_gemm(
    const unsigned short* __restrict__ ctx,
    const unsigned short* __restrict__ wo,
    const float* __restrict__ bo,
    float* __restrict__ out) {
  __shared__ __align__(16) unsigned short bsh[2][64 * 32];

  const int lane = threadIdx.x & 31;
  const int wave = threadIdx.x >> 5;
  const int lo = lane & 15, hi = lane >> 4;
  const int m0 = (blockIdx.x * 8 + wave) * 32;
  const int nb = blockIdx.y * 64;

  const unsigned short* arow0 = ctx + (size_t)(m0 + lo) * D_MODEL;
  const unsigned short* arow1 = ctx + (size_t)(m0 + 16 + lo) * D_MODEL;

  if (wave == 0)
    tdm_load_w_tile(wo + (size_t)nb * D_MODEL, (unsigned)(uintptr_t)&bsh[0][0]);

  v8f acc[2][4] = {};
  for (int k = 0; k < D_MODEL; k += 32) {
    const int cur = (k >> 5) & 1;
    if (wave == 0) __builtin_amdgcn_s_wait_tensorcnt(0);
    __syncthreads();
    if (wave == 0 && (k + 32) < D_MODEL)
      tdm_load_w_tile(wo + (size_t)nb * D_MODEL + (k + 32),
                      (unsigned)(uintptr_t)&bsh[cur ^ 1][0]);

    __builtin_prefetch(arow0 + k + 64, 0, 1);
    __builtin_prefetch(arow1 + k + 64, 0, 1);
    v16bf a0 = load_a_frag(arow0, k, hi);
    v16bf a1 = load_a_frag(arow1, k, hi);
#pragma unroll
    for (int j = 0; j < 4; ++j) {
      const unsigned short* bcol = &bsh[cur][(j * 16 + lo) * 32 + 16 * hi];
      FragU bf;
      bf.q[0] = *(const uint4*)(bcol);
      bf.q[1] = *(const uint4*)(bcol + 8);
      acc[0][j] = wmma_bf16(a0, bf.v, acc[0][j]);
      acc[1][j] = wmma_bf16(a1, bf.v, acc[1][j]);
    }
  }

#pragma unroll
  for (int mi = 0; mi < 2; ++mi) {
#pragma unroll
    for (int j = 0; j < 4; ++j) {
      int n = nb + j * 16 + lo;
      float bias = bo[n];
#pragma unroll
      for (int r = 0; r < 8; ++r) {
        int m = m0 + mi * 16 + r + 8 * hi;
        out[(size_t)m * D_MODEL + n] = acc[mi][j][r] + bias;
      }
    }
  }
}

extern "C" void kernel_launch(void* const* d_in, const int* in_sizes, int n_in,
                              void* d_out, int out_size, void* d_ws, size_t ws_size,
                              hipStream_t stream) {
  const float* x  = (const float*)d_in[0];
  const float* Wq = (const float*)d_in[1];
  const float* Wk = (const float*)d_in[2];
  const float* Wv = (const float*)d_in[3];
  const float* Wo = (const float*)d_in[4];
  const float* bo = (const float*)d_in[5];

  unsigned short* ws  = (unsigned short*)d_ws;
  unsigned short* xh  = ws + OFF_XH;
  unsigned short* wq  = ws + OFF_WQ;
  unsigned short* wk  = ws + OFF_WK;
  unsigned short* wv  = ws + OFF_WV;
  unsigned short* wo  = ws + OFF_WO;
  unsigned short* qh  = ws + OFF_Q;
  unsigned short* kh  = ws + OFF_K;
  unsigned short* vt  = ws + OFF_VT;
  unsigned short* ctx = ws + OFF_CTX;

  convert_bf16<<<2048, 256, 0, stream>>>(x,  xh, M_TOK * D_MODEL);
  convert_bf16<<<512,  256, 0, stream>>>(Wq, wq, D_MODEL * D_MODEL);
  convert_bf16<<<512,  256, 0, stream>>>(Wk, wk, D_MODEL * D_MODEL);
  convert_bf16<<<512,  256, 0, stream>>>(Wv, wv, D_MODEL * D_MODEL);
  convert_bf16<<<512,  256, 0, stream>>>(Wo, wo, D_MODEL * D_MODEL);

  qkv_gemm<<<dim3(16, 16, 3), 256, 0, stream>>>(xh, wq, wk, wv, qh, kh, vt);
  attn    <<<dim3(32, 16, 2), 128, 0, stream>>>(qh, kh, vt, ctx);
  out_gemm<<<dim3(16, 16),    256, 0, stream>>>(ctx, wo, bo, (float*)d_out);
}
